// Decoder_3496103379512
// MI455X (gfx1250) — matrix operand order repaired and checked by reading, not compiled, not run on hardware
//
#include <hip/hip_runtime.h>
#include <hip/hip_bf16.h>
#include <stddef.h>

// ---------------------------------------------------------------------------
// Tacotron2-style decoder on MI455X (gfx1250), wave32 + WMMA bf16.
//
// Key simplification: LSTM h/c are reset to zero every step in the reference,
// so the Whh matmuls and forget gate are dead. Each step only needs
//   g = x @ Wih^T + bih + bhh   (i, g, o gate blocks = 3072 of 4096 cols)
//   h = sigmoid(o) * tanh( sigmoid(i) * tanh(g) )
// Prenet + proc_enc are step-invariant -> hoisted to batched WMMA GEMMs.
// Weights are converted once to bf16 (fits in L2: ~15 MB) and streamed into
// V_WMMA_F32_16X16X32_BF16 with f32 accumulation.
// ---------------------------------------------------------------------------

typedef unsigned short u16;
typedef __attribute__((ext_vector_type(16))) __bf16 v16bf;
typedef __attribute__((ext_vector_type(8)))  float  v8f;

__device__ __forceinline__ __bf16 bits2bf(u16 v) {
    return __builtin_bit_cast(__bf16, v);
}
__device__ __forceinline__ u16 f2bf(float f) {
    unsigned u = __float_as_uint(f);
    u += 0x7FFFu + ((u >> 16) & 1u);     // round-to-nearest-even
    return (u16)(u >> 16);
}
__device__ __forceinline__ float sigf(float x) {
    return 1.0f / (1.0f + __expf(-x));
}

// A fragment: 16x32 bf16, row-major source with leading dim lda.
// ISA 7.12.2: lane m=l%16, half h=l/16; VGPR j(0..3): K=h*8+2j,+1; j(4..7): +16.
__device__ __forceinline__ v16bf load_a(const u16* src, int lda, int lane) {
    int m = lane & 15, h = lane >> 4;
    const u16* row = src + (size_t)m * lda;
    v16bf a;
#pragma unroll
    for (int j = 0; j < 4; ++j) {
        int k = h * 8 + 2 * j;
        a[2 * j]     = bits2bf(row[k]);
        a[2 * j + 1] = bits2bf(row[k + 1]);
        a[8 + 2 * j]     = bits2bf(row[16 + k]);
        a[8 + 2 * j + 1] = bits2bf(row[16 + k + 1]);
    }
    return a;
}

// B fragment for Y = X @ W^T with W row-major [N,K]: B[k,n] = W[n,k].
// Lane n=l%16, half h=l/16: elems 0..7 K=h*8..h*8+7, elems 8..15 K=16+h*8..16+h*8+7.
__device__ __forceinline__ v16bf load_b(const u16* Wrow0, int ldw, int lane) {
    int n = lane & 15, h = lane >> 4;
    const u16* p = Wrow0 + (size_t)n * ldw + h * 8;
    v16bf b;
#pragma unroll
    for (int j = 0; j < 8; ++j) { b[j] = bits2bf(p[j]); b[8 + j] = bits2bf(p[16 + j]); }
    return b;
}

#define WMMA_BF16(a, b, c) \
    __builtin_amdgcn_wmma_f32_16x16x32_bf16(false, (a), false, (b), (short)0, (c), false, false)

// ---------------------------------------------------------------------------
// Generic tiled WMMA GEMM: out = act(A[M,K] @ W[N,K]^T + bias)
// One 16x16 tile per wave. blockDim must be 256 (8 full waves -> EXEC all 1s).
// ---------------------------------------------------------------------------
template <int RELU, int OUT_BF16>
__global__ __launch_bounds__(256) void wmma_gemm_kernel(
    const u16* __restrict__ A, int lda, const u16* __restrict__ W, int ldw,
    const float* __restrict__ bias, float* __restrict__ outF,
    u16* __restrict__ outB, int ldo, int Mtiles, int Ntiles, int K)
{
    int lane = threadIdx.x & 31;
    int wid  = threadIdx.x >> 5;
    int tile = blockIdx.x * 8 + wid;
    if (tile >= Mtiles * Ntiles) return;          // uniform per wave
    int mt = tile / Ntiles, nt = tile - mt * Ntiles;
    const u16* Abase = A + (size_t)(mt * 16) * lda;
    const u16* Wbase = W + (size_t)(nt * 16) * ldw;
    v8f acc = 0.0f;
    for (int k = 0; k < K; k += 32) {
        __builtin_prefetch(Wbase + k + 128, 0, 1);
        v16bf a = load_a(Abase + k, lda, lane);
        v16bf b = load_b(Wbase + k, ldw, lane);
        acc = WMMA_BF16(a, b, acc);
    }
    int n = lane & 15, h = lane >> 4;
    int gn = nt * 16 + n;
    float bb = bias ? bias[gn] : 0.0f;
#pragma unroll
    for (int r = 0; r < 8; ++r) {
        int gm  = mt * 16 + r + 8 * h;
        float v = acc[r] + bb;
        if (RELU) v = v > 0.0f ? v : 0.0f;
        if (OUT_BF16) outB[(size_t)gm * ldo + gn] = f2bf(v);
        else          outF[(size_t)gm * ldo + gn] = v;
    }
}

// ---------------------------------------------------------------------------
// Fused degenerate-LSTM cell (h=c=0):  3 gate GEMM tiles per wave sharing A.
//   K1: A = [pre_t(256) | ctx(512)] built in LDS;  K2: A = h1 bf16.
// Wih rows: i at [0,1024), g at [2048,3072), o at [3072,4096).
// grid=16, block=256 -> 128 waves = 2 Mtiles x 64 Ntiles.
// ---------------------------------------------------------------------------
__global__ __launch_bounds__(256) void lstm_cell_kernel(
    const u16* __restrict__ A, const float* __restrict__ ctx,
    const u16* __restrict__ Wih, int Kdim,
    const float* __restrict__ bih, const float* __restrict__ bhh,
    u16* __restrict__ h_bf, float* __restrict__ h_f)
{
    __shared__ u16 Ash[32 * 1024];
    int total = 32 * Kdim;
    for (int idx = threadIdx.x; idx < total; idx += blockDim.x) {
        int m = idx / Kdim, k = idx - m * Kdim;
        u16 v;
        if (ctx) v = (k < 256) ? A[m * 256 + k] : f2bf(ctx[m * 512 + (k - 256)]);
        else     v = A[idx];
        Ash[idx] = v;
    }
    __syncthreads();

    int lane = threadIdx.x & 31;
    int wid  = threadIdx.x >> 5;
    int gw   = blockIdx.x * 8 + wid;      // 0..127
    int mt   = gw & 1;
    int nt   = gw >> 1;                   // 0..63
    const u16* Wi = Wih + (size_t)(nt * 16) * Kdim;
    const u16* Wg = Wih + (size_t)(2048 + nt * 16) * Kdim;
    const u16* Wo = Wih + (size_t)(3072 + nt * 16) * Kdim;
    const u16* Ab = Ash + (size_t)(mt * 16) * Kdim;

    v8f ci = 0.0f, cg = 0.0f, co = 0.0f;
    for (int k = 0; k < Kdim; k += 32) {
        __builtin_prefetch(Wi + k + 256, 0, 1);
        v16bf a = load_a(Ab + k, Kdim, lane);
        ci = WMMA_BF16(a, load_b(Wi + k, Kdim, lane), ci);
        cg = WMMA_BF16(a, load_b(Wg + k, Kdim, lane), cg);
        co = WMMA_BF16(a, load_b(Wo + k, Kdim, lane), co);
    }

    int n = lane & 15, h = lane >> 4;
    int col = nt * 16 + n;
    float bi = bih[col]        + bhh[col];
    float bg = bih[2048 + col] + bhh[2048 + col];
    float bo = bih[3072 + col] + bhh[3072 + col];
#pragma unroll
    for (int r = 0; r < 8; ++r) {
        float iv = ci[r] + bi, gv = cg[r] + bg, ov = co[r] + bo;
        float cn = sigf(iv) * tanhf(gv);
        float hv = sigf(ov) * tanhf(cn);
        int gm = mt * 16 + r + 8 * h;
        h_bf[(size_t)gm * 1024 + col] = f2bf(hv);
        if (h_f) h_f[(size_t)gm * 1024 + col] = hv;
    }
}

// ---------------------------------------------------------------------------
// Location conv: conv_out[b,t,c] = sum_k aw_cum[b,t+k-15] * W_conv[c,0,k]
// Output bf16 laid out [b*256+t, 32] = A-matrix for the pa GEMM.
// ---------------------------------------------------------------------------
__global__ void conv_kernel(const float* __restrict__ aw_cum,
                            const float* __restrict__ W_conv,
                            u16* __restrict__ conv_bf)
{
    int idx = blockIdx.x * blockDim.x + threadIdx.x;
    if (idx >= 32 * 256 * 32) return;
    int c = idx & 31;
    int t = (idx >> 5) & 255;
    int b = idx >> 13;
    float s = 0.0f;
#pragma unroll
    for (int k = 0; k < 31; ++k) {
        int tt = t + k - 15;
        if (tt >= 0 && tt < 256) s += aw_cum[b * 256 + tt] * W_conv[c * 31 + k];
    }
    conv_bf[idx] = f2bf(s);
}

// ---------------------------------------------------------------------------
// Energies: pa tile via WMMA (K=32), fused + proc_enc + lin, tanh, dot W_e,
// reduced over f with LDS ds_add_f32 atomics. 16 rows per block, 8 waves.
// ---------------------------------------------------------------------------
__global__ __launch_bounds__(256) void energies_kernel(
    const u16* __restrict__ conv_bf, const u16* __restrict__ wloc_bf,
    const float* __restrict__ proc_enc, const float* __restrict__ lin,
    const float* __restrict__ W_e, float* __restrict__ energies)
{
    __shared__ float earr[16];
    int lane = threadIdx.x & 31;
    int wid  = threadIdx.x >> 5;
    int r0   = blockIdx.x * 16;
    if (threadIdx.x < 16) earr[threadIdx.x] = 0.0f;
    __syncthreads();

    v16bf a = load_a(conv_bf + (size_t)r0 * 32, 32, lane);
    v16bf b = load_b(wloc_bf + (size_t)(wid * 16) * 32, 32, lane);
    v8f acc = 0.0f;
    acc = WMMA_BF16(a, b, acc);

    int n = lane & 15, h = lane >> 4;
    int f = wid * 16 + n;
    float we = W_e[f];
#pragma unroll
    for (int r = 0; r < 8; ++r) {
        int m = r + 8 * h;
        int row = r0 + m;
        int bidx = row >> 8;
        float v = acc[r] + proc_enc[(size_t)row * 128 + f] + lin[bidx * 128 + f];
        atomicAdd(&earr[m], tanhf(v) * we);      // ds_add_f32
    }
    __syncthreads();
    if (threadIdx.x < 16) energies[r0 + threadIdx.x] = earr[threadIdx.x];
}

// ---------------------------------------------------------------------------
// Per-batch softmax over T_enc + aw_cum update + ctx = aw @ inputs[b].
// ---------------------------------------------------------------------------
__global__ __launch_bounds__(256) void softmax_ctx_kernel(
    const float* __restrict__ energies, const float* __restrict__ inputs,
    float* __restrict__ aw_cum, float* __restrict__ ctx)
{
    __shared__ float sh[256];
    __shared__ float awl[256];
    int b = blockIdx.x, tid = threadIdx.x;
    float e = energies[b * 256 + tid];
    sh[tid] = e; __syncthreads();
    for (int s = 128; s > 0; s >>= 1) {
        if (tid < s) sh[tid] = fmaxf(sh[tid], sh[tid + s]);
        __syncthreads();
    }
    float mx = sh[0]; __syncthreads();
    float ex = __expf(e - mx);
    sh[tid] = ex; __syncthreads();
    for (int s = 128; s > 0; s >>= 1) {
        if (tid < s) sh[tid] += sh[tid + s];
        __syncthreads();
    }
    float aw = ex / sh[0];
    awl[tid] = aw;
    aw_cum[b * 256 + tid] += aw;
    __syncthreads();

    const float* inb = inputs + (size_t)b * 256 * 512;
    float a0 = 0.0f, a1 = 0.0f;
    for (int t = 0; t < 256; ++t) {
        float w = awl[t];
        a0 += w * inb[(size_t)t * 512 + tid];
        a1 += w * inb[(size_t)t * 512 + tid + 256];
    }
    ctx[b * 512 + tid]       = a0;
    ctx[b * 512 + tid + 256] = a1;
}

// ---------------------------------------------------------------------------
// Output frame: out[b,t,:] = [h2 | ctx] @ W_out^T + b_out   (32x80, K=1536)
// ---------------------------------------------------------------------------
__global__ void frame_kernel(const float* __restrict__ h2,
                             const float* __restrict__ ctx,
                             const float* __restrict__ W_out,
                             const float* __restrict__ b_out,
                             float* __restrict__ out, int t)
{
    int idx = blockIdx.x * blockDim.x + threadIdx.x;
    if (idx >= 32 * 80) return;
    int b = idx / 80, j = idx - b * 80;
    const float* wr = W_out + (size_t)j * 1536;
    const float* hh = h2 + b * 1024;
    const float* cc = ctx + b * 512;
    float acc = b_out[j];
    for (int k = 0; k < 1024; ++k) acc += hh[k] * wr[k];
    for (int k = 0; k < 512;  ++k) acc += cc[k] * wr[1024 + k];
    out[(size_t)b * 256 * 80 + (size_t)t * 80 + j] = acc;
}

// ----------------------------- prep kernels --------------------------------
__global__ void f32_to_bf16_kernel(const float* __restrict__ in,
                                   u16* __restrict__ out, int n)
{
    int i = blockIdx.x * blockDim.x + threadIdx.x;
    if (i < n) out[i] = f2bf(in[i]);
}

__global__ void zero_f32_kernel(float* p, int n)
{
    int i = blockIdx.x * blockDim.x + threadIdx.x;
    if (i < n) p[i] = 0.0f;
}

// Pad W_pre1 [256,80] -> bf16 [256,96] (zero K-pad so K%32==0).
__global__ void pad_wpre1_kernel(const float* __restrict__ W, u16* __restrict__ out)
{
    int idx = blockIdx.x * blockDim.x + threadIdx.x;
    if (idx >= 256 * 96) return;
    int n = idx / 96, k = idx - n * 96;
    out[idx] = f2bf(k < 80 ? W[n * 80 + k] : 0.0f);
}

// Teacher-forcing input X[(t*32+b), 96]: ch<80 & t>0 -> mels[b,ch,t-1], else 0.
__global__ void build_x_kernel(const float* __restrict__ mels, u16* __restrict__ X)
{
    int idx = blockIdx.x * blockDim.x + threadIdx.x;
    if (idx >= 8192 * 96) return;
    int row = idx / 96, ch = idx - row * 96;
    int t = row >> 5, b = row & 31;
    float v = 0.0f;
    if (ch < 80 && t > 0) v = mels[((size_t)b * 80 + ch) * 256 + (t - 1)];
    X[idx] = f2bf(v);
}

// ---------------------------------------------------------------------------
extern "C" void kernel_launch(void* const* d_in, const int* in_sizes, int n_in,
                              void* d_out, int out_size, void* d_ws, size_t ws_size,
                              hipStream_t stream)
{
    const float* inputs = (const float*)d_in[0];
    const float* mels   = (const float*)d_in[1];
    const float* W_pre1 = (const float*)d_in[2];
    const float* b_pre1 = (const float*)d_in[3];
    const float* W_pre2 = (const float*)d_in[4];
    const float* b_pre2 = (const float*)d_in[5];
    const float* W_enc  = (const float*)d_in[6];
    const float* W_lstm = (const float*)d_in[7];
    const float* W_conv = (const float*)d_in[8];
    const float* W_loc  = (const float*)d_in[9];
    const float* W_e    = (const float*)d_in[10];
    const float* Wih0   = (const float*)d_in[11];
    // d_in[12] Whh0 unused: h==0 every step
    const float* bih0   = (const float*)d_in[13];
    const float* bhh0   = (const float*)d_in[14];
    const float* Wih1   = (const float*)d_in[15];
    // d_in[16] Whh1 unused
    const float* bih1   = (const float*)d_in[17];
    const float* bhh1   = (const float*)d_in[18];
    const float* W_out  = (const float*)d_in[19];
    const float* b_out  = (const float*)d_in[20];
    float* out = (float*)d_out;

    // workspace carve (256B aligned)
    char* p = (char*)d_ws;
    auto carve = [&](size_t bytes) -> char* {
        char* r = p;
        p += (bytes + 255) & ~(size_t)255;
        return r;
    };
    u16*   wih0_bf  = (u16*)carve((size_t)4096 * 768 * 2);
    u16*   wih1_bf  = (u16*)carve((size_t)4096 * 1024 * 2);
    u16*   wlstm_bf = (u16*)carve((size_t)128 * 1024 * 2);
    u16*   wloc_bf  = (u16*)carve((size_t)128 * 32 * 2);
    u16*   wpre1_bf = (u16*)carve((size_t)256 * 96 * 2);
    u16*   wpre2_bf = (u16*)carve((size_t)256 * 256 * 2);
    u16*   wenc_bf  = (u16*)carve((size_t)128 * 512 * 2);
    u16*   in_bf    = (u16*)carve((size_t)8192 * 512 * 2);
    u16*   X_bf     = (u16*)carve((size_t)8192 * 96 * 2);
    u16*   H1p_bf   = (u16*)carve((size_t)8192 * 256 * 2);
    u16*   pre_bf   = (u16*)carve((size_t)8192 * 256 * 2);
    float* proc_enc = (float*)carve((size_t)8192 * 128 * 4);
    float* ctx_f    = (float*)carve((size_t)32 * 512 * 4);
    float* awcum_f  = (float*)carve((size_t)32 * 256 * 4);
    u16*   h1_bf    = (u16*)carve((size_t)32 * 1024 * 2);
    float* h2_f     = (float*)carve((size_t)32 * 1024 * 4);
    u16*   h2_bf    = (u16*)carve((size_t)32 * 1024 * 2);
    float* lin_f    = (float*)carve((size_t)32 * 128 * 4);
    u16*   conv_bf  = (u16*)carve((size_t)8192 * 32 * 2);
    float* energ_f  = (float*)carve((size_t)8192 * 4);

    const int TPB = 256;
    auto g = [](int n) { return (n + 255) / 256; };

    // ---- init recurrent state ----
    zero_f32_kernel<<<g(32 * 512), TPB, 0, stream>>>(ctx_f, 32 * 512);
    zero_f32_kernel<<<g(32 * 256), TPB, 0, stream>>>(awcum_f, 32 * 256);

    // ---- weight / input conversions to bf16 ----
    f32_to_bf16_kernel<<<g(4096 * 768),  TPB, 0, stream>>>(Wih0,   wih0_bf,  4096 * 768);
    f32_to_bf16_kernel<<<g(4096 * 1024), TPB, 0, stream>>>(Wih1,   wih1_bf,  4096 * 1024);
    f32_to_bf16_kernel<<<g(128 * 1024),  TPB, 0, stream>>>(W_lstm, wlstm_bf, 128 * 1024);
    f32_to_bf16_kernel<<<g(128 * 32),    TPB, 0, stream>>>(W_loc,  wloc_bf,  128 * 32);
    f32_to_bf16_kernel<<<g(256 * 256),   TPB, 0, stream>>>(W_pre2, wpre2_bf, 256 * 256);
    f32_to_bf16_kernel<<<g(128 * 512),   TPB, 0, stream>>>(W_enc,  wenc_bf,  128 * 512);
    f32_to_bf16_kernel<<<g(8192 * 512),  TPB, 0, stream>>>(inputs, in_bf,    8192 * 512);
    pad_wpre1_kernel<<<g(256 * 96), TPB, 0, stream>>>(W_pre1, wpre1_bf);
    build_x_kernel<<<g(8192 * 96), TPB, 0, stream>>>(mels, X_bf);

    // ---- hoisted batched GEMMs (WMMA) ----
    // prenet layer 1: relu(X[8192,96] @ W_pre1^T + b) -> bf16 [8192,256]
    wmma_gemm_kernel<1, 1><<<1024, TPB, 0, stream>>>(
        X_bf, 96, wpre1_bf, 96, b_pre1, nullptr, H1p_bf, 256, 512, 16, 96);
    // prenet layer 2: relu(H1p @ W_pre2^T + b) -> bf16 [8192,256]
    wmma_gemm_kernel<1, 1><<<1024, TPB, 0, stream>>>(
        H1p_bf, 256, wpre2_bf, 256, b_pre2, nullptr, pre_bf, 256, 512, 16, 256);
    // proc_enc = inputs @ W_enc^T -> f32 [8192,128]
    wmma_gemm_kernel<0, 0><<<512, TPB, 0, stream>>>(
        in_bf, 512, wenc_bf, 512, nullptr, proc_enc, nullptr, 128, 512, 8, 512);

    // ---- sequential decode: 256 steps ----
    for (int t = 0; t < 256; ++t) {
        // LSTM0: A = [pre_t | ctx] (K=768), fused i/g/o -> h1 bf16
        lstm_cell_kernel<<<16, TPB, 0, stream>>>(
            pre_bf + (size_t)t * 32 * 256, ctx_f, wih0_bf, 768,
            bih0, bhh0, h1_bf, nullptr);
        // LSTM1: A = h1 (K=1024) -> h2 bf16 + f32
        lstm_cell_kernel<<<16, TPB, 0, stream>>>(
            h1_bf, nullptr, wih1_bf, 1024, bih1, bhh1, h2_bf, h2_f);
        // lin = h2 @ W_lstm^T -> f32 [32,128]   (16 tiles)
        wmma_gemm_kernel<0, 0><<<2, TPB, 0, stream>>>(
            h2_bf, 1024, wlstm_bf, 1024, nullptr, lin_f, nullptr, 128, 2, 8, 1024);
        // location conv of cumulative attention
        conv_kernel<<<1024, TPB, 0, stream>>>(awcum_f, W_conv, conv_bf);
        // energies: pa WMMA + tanh + dot W_e
        energies_kernel<<<512, TPB, 0, stream>>>(
            conv_bf, wloc_bf, proc_enc, lin_f, W_e, energ_f);
        // softmax + aw_cum update + new context
        softmax_ctx_kernel<<<32, TPB, 0, stream>>>(energ_f, inputs, awcum_f, ctx_f);
        // output frame
        frame_kernel<<<10, TPB, 0, stream>>>(h2_f, ctx_f, W_out, b_out, out, t);
    }
}
